// AttentionUnit_5317169512783
// MI455X (gfx1250) — matrix operand order repaired and checked by reading, not compiled, not run on hardware
//
#include <hip/hip_runtime.h>
#include <hip/hip_bf16.h>
#include <cstddef>
#include <cstdint>

// ---------------------------------------------------------------------------
// Problem constants (from reference): B=16, Cin=131, Cmid=256, Cout=128, N=2048
// ---------------------------------------------------------------------------
#define BSZ   16
#define CIN   131
#define CMID  256
#define COUT  128
#define NPTS  2048
#define EPSV  1e-5f
#define NKB   (NPTS / 32)          // 64 key blocks of 32 rows

typedef __attribute__((ext_vector_type(16))) __bf16 v16bf;
typedef __attribute__((ext_vector_type(8)))  float  v8f;

union BFrag {            // 16 bf16 = 32 bytes = 8 VGPRs (one WMMA A/B operand)
    v16bf v;
    uint4 q[2];
};

__device__ __forceinline__ unsigned short f2bf(float f) {
    unsigned u = __float_as_uint(f);
    u = (u + 0x7FFFu + ((u >> 16) & 1u)) >> 16;   // round-to-nearest-even
    return (unsigned short)u;
}
__device__ __forceinline__ float bf2f(unsigned short h) {
    return __uint_as_float(((unsigned)h) << 16);
}

// ---------------------------------------------------------------------------
// CDNA5 async global->LDS DMA (ASYNCcnt path), with portable fallback.
// Probe-derived signature: (AS1 int4vec* src, AS3 int4vec* dst, Imm, Imm).
// ---------------------------------------------------------------------------
#if defined(__has_builtin)
#if __has_builtin(__builtin_amdgcn_global_load_async_to_lds_b128)
#define HAVE_ASYNC_LDS 1
#endif
#endif

#ifdef HAVE_ASYNC_LDS
typedef int v4i __attribute__((vector_size(16)));   // GCC vector, not HIP int4
typedef __attribute__((address_space(1))) v4i gv4i_t;
typedef __attribute__((address_space(3))) v4i lv4i_t;
__device__ __forceinline__ void copy16B_to_lds(const unsigned short* g,
                                               unsigned short* l) {
    __builtin_amdgcn_global_load_async_to_lds_b128(
        (gv4i_t*)g, (lv4i_t*)l, 0, 0);
}
#define ASYNC_WAIT() asm volatile("s_wait_asynccnt 0" ::: "memory")
#else
__device__ __forceinline__ void copy16B_to_lds(const unsigned short* g,
                                               unsigned short* l) {
    *(uint4*)l = *(const uint4*)g;
}
#define ASYNC_WAIT() do {} while (0)
#endif

// ---------------------------------------------------------------------------
// Kernel 1: Q/K/V 1x1-conv + BN + ReLU, f32 in -> bf16 out.
// Q,K stored [b][n][c] (row = 256 contiguous bf16), V stored [b][c][n].
// Grid: B * N/16 workgroups of 256 threads; thread t owns channel c=t.
// ---------------------------------------------------------------------------
__device__ __forceinline__ void proj_bn_relu(const float* __restrict__ W,
                                             const float* __restrict__ bn,
                                             const float* __restrict__ xl,
                                             int c, float acc[16]) {
    #pragma unroll
    for (int nn = 0; nn < 16; ++nn) acc[nn] = 0.f;
    for (int ci = 0; ci < CIN; ++ci) {
        float w = W[c * CIN + ci];
        #pragma unroll
        for (int nn = 0; nn < 16; ++nn) acc[nn] = fmaf(w, xl[ci * 16 + nn], acc[nn]);
    }
    float g  = bn[c];
    float be = bn[CMID + c];
    float mu = bn[2 * CMID + c];
    float va = bn[3 * CMID + c];
    float sc = g * rsqrtf(va + EPSV);
    #pragma unroll
    for (int nn = 0; nn < 16; ++nn) {
        float y = (acc[nn] - mu) * sc + be;
        acc[nn] = fmaxf(y, 0.f);
    }
}

__global__ __launch_bounds__(256) void qkv_proj_kernel(
    const float* __restrict__ x,
    const float* __restrict__ Wq, const float* __restrict__ Wk,
    const float* __restrict__ Wv,
    const float* __restrict__ bnq, const float* __restrict__ bnk,
    const float* __restrict__ bnv,
    unsigned short* __restrict__ qt, unsigned short* __restrict__ kt,
    unsigned short* __restrict__ vt)
{
    __shared__ float xl[CIN * 16];
    const int b  = blockIdx.x >> 7;          // 128 tiles per batch
    const int n0 = (blockIdx.x & 127) * 16;
    const int t  = threadIdx.x;

    for (int i = t; i < CIN * 16; i += 256) {
        int ci = i >> 4, nn = i & 15;
        xl[i] = x[((size_t)b * CIN + ci) * NPTS + n0 + nn];
    }
    __syncthreads();

    const int c = t;                          // 0..255
    float acc[16];

    proj_bn_relu(Wq, bnq, xl, c, acc);
    #pragma unroll
    for (int nn = 0; nn < 16; ++nn)
        qt[((size_t)(b * NPTS + n0 + nn)) * CMID + c] = f2bf(acc[nn]);

    proj_bn_relu(Wk, bnk, xl, c, acc);
    #pragma unroll
    for (int nn = 0; nn < 16; ++nn)
        kt[((size_t)(b * NPTS + n0 + nn)) * CMID + c] = f2bf(acc[nn]);

    proj_bn_relu(Wv, bnv, xl, c, acc);
    #pragma unroll
    for (int nn = 0; nn < 16; ++nn)
        vt[((size_t)(b * CMID + c)) * NPTS + n0 + nn] = f2bf(acc[nn]);
}

// ---------------------------------------------------------------------------
// Kernel 2: fused flash attention on WMMA bf16, double-buffered K/V in LDS.
// Grid: B*16 workgroups x 256 threads (8 waves). Wave w owns query tile
// n0 = (grp*8+w)*16. Loop over 64 key blocks of 32 rows.
// Steady state per block: issue async DMA for block mb+1, compute 32 WMMAs
// + online softmax on block mb, then s_wait_asynccnt + one barrier.
// ---------------------------------------------------------------------------
__global__ __launch_bounds__(256) void flash_attn_kernel(
    const unsigned short* __restrict__ qt,
    const unsigned short* __restrict__ kt,
    const unsigned short* __restrict__ vt,
    unsigned short* __restrict__ attn)
{
    __shared__ __align__(32) unsigned short k_lds[2][32 * CMID];   // 32 KB
    __shared__ __align__(32) unsigned short v_lds[2][CMID * 32];   // 32 KB
    __shared__ __align__(32) unsigned short p_lds[8][16 * 32];     //  8 KB

    const int b    = blockIdx.x >> 4;
    const int grp  = blockIdx.x & 15;
    const int t    = threadIdx.x;
    const int wv   = t >> 5;
    const int lane = t & 31;
    const int lo   = lane & 15;
    const int hi   = lane >> 4;
    const int n0   = (grp * 8 + wv) * 16;

    // Per-thread staging addresses (row/col assignment is mb-invariant).
    const int krow = t >> 3, kcol = (t & 7) * 32;
    const unsigned short* gk_base =
        kt + ((size_t)(b * NPTS + krow)) * CMID + kcol;       // + m0*CMID
    const unsigned short* gv_base =
        vt + ((size_t)(b * CMID + t)) * NPTS;                 // + m0

    // ---- stage one 32-row K/V block into LDS buffer `buf`
    auto stage = [&](int mb, int buf) {
        const int m0 = mb * 32;
        const unsigned short* gk = gk_base + (size_t)m0 * CMID;
        unsigned short*       lk = k_lds[buf] + krow * CMID + kcol;
        const unsigned short* gv = gv_base + m0;
        unsigned short*       lv = v_lds[buf] + t * 32;
        #pragma unroll
        for (int j = 0; j < 4; ++j) {
            copy16B_to_lds(gk + j * 8, lk + j * 8);
            copy16B_to_lds(gv + j * 8, lv + j * 8);
        }
    };

    // ---- Preload Q A-fragments (16x32 bf16 each), K dim = 256 in 8 chunks.
    // ISA 16-bit A layout: lane = row (M = lo), lane-half selects K-subset:
    // half 0 -> K {0..7,16..23}, half 1 -> K {8..15,24..31}: two 16B chunks.
    BFrag qa[8];
    const unsigned short* qrow = qt + ((size_t)(b * NPTS + n0 + lo)) * CMID;
    #pragma unroll
    for (int kk = 0; kk < 8; ++kk) {
        int c0 = kk * 32 + hi * 8;
        qa[kk].q[0] = *(const uint4*)(qrow + c0);
        qa[kk].q[1] = *(const uint4*)(qrow + c0 + 16);
    }

    const v8f vzero = {0.f, 0.f, 0.f, 0.f, 0.f, 0.f, 0.f, 0.f};
    v8f O[16];                       // 16 c-tiles x (16n x 16c) f32 accum
    #pragma unroll
    for (int i = 0; i < 16; ++i) O[i] = vzero;

    float mrow[8], lrow[8];          // per-lane row stats (row = r + 8*hi)
    #pragma unroll
    for (int r = 0; r < 8; ++r) { mrow[r] = -3.0e38f; lrow[r] = 0.f; }

    stage(0, 0);
    ASYNC_WAIT();
    __syncthreads();

    for (int mb = 0; mb < NKB; ++mb) {
        const int cur = mb & 1;
        if (mb + 1 < NKB) stage(mb + 1, cur ^ 1);   // overlap DMA with compute

        const unsigned short* kb = k_lds[cur];
        const unsigned short* vb = v_lds[cur];

        // ---- scores: two 16x16 tiles (m halves), K=256 via 8 WMMAs each.
        // B 32x16 layout: lane = column (m = th*16+lo), K split like A (8h.., 16+8h..).
        v8f s[2];
        #pragma unroll
        for (int th = 0; th < 2; ++th) {
            v8f acc = vzero;
            #pragma unroll
            for (int kk = 0; kk < 8; ++kk) {
                BFrag bk;
                const unsigned short* kr =
                    kb + (th * 16 + lo) * CMID + kk * 32 + hi * 8;
                bk.q[0] = *(const uint4*)(kr);
                bk.q[1] = *(const uint4*)(kr + 16);
                acc = __builtin_amdgcn_wmma_f32_16x16x32_bf16(
                          false, qa[kk].v, false, bk.v, (short)0, acc, false, false);
            }
            s[th] = acc;
        }

        // ---- online softmax. D layout: lane = col, row r lives in the 16
        // lanes of this half -> xor masks 1,2,4,8 reduce within the half.
        float alpha[8];
        #pragma unroll
        for (int r = 0; r < 8; ++r) {
            float mx = fmaxf(s[0][r], s[1][r]);
            mx = fmaxf(mx, __shfl_xor(mx, 1, 32));
            mx = fmaxf(mx, __shfl_xor(mx, 2, 32));
            mx = fmaxf(mx, __shfl_xor(mx, 4, 32));
            mx = fmaxf(mx, __shfl_xor(mx, 8, 32));
            float nm = fmaxf(mrow[r], mx);
            alpha[r] = __expf(mrow[r] - nm);
            mrow[r]  = nm;
            float p0 = __expf(s[0][r] - nm);
            float p1 = __expf(s[1][r] - nm);
            s[0][r] = p0; s[1][r] = p1;
            float ps = p0 + p1;
            ps += __shfl_xor(ps, 1, 32);
            ps += __shfl_xor(ps, 2, 32);
            ps += __shfl_xor(ps, 4, 32);
            ps += __shfl_xor(ps, 8, 32);
            lrow[r] = lrow[r] * alpha[r] + ps;
        }
        #pragma unroll
        for (int ct = 0; ct < 16; ++ct) {
            #pragma unroll
            for (int r = 0; r < 8; ++r) O[ct][r] *= alpha[r];
        }

        // ---- P: D-layout -> A-layout transpose via per-wave LDS scratch.
        unsigned short* pl = p_lds[wv];
        #pragma unroll
        for (int th = 0; th < 2; ++th) {
            #pragma unroll
            for (int r = 0; r < 8; ++r)
                pl[(r + 8 * hi) * 32 + th * 16 + lo] = f2bf(s[th][r]);
        }
        asm volatile("s_wait_dscnt 0" ::: "memory"); // in-wave DS RAW fence

        BFrag pa;                   // A 16x32 fragment of probabilities
        {
            const unsigned short* pr = pl + lo * 32 + hi * 8;
            pa.q[0] = *(const uint4*)(pr);
            pa.q[1] = *(const uint4*)(pr + 16);
        }

        // ---- PV: 16 c-tiles, one K=32 WMMA each.
        #pragma unroll
        for (int ct = 0; ct < 16; ++ct) {
            BFrag bv;
            const unsigned short* vr = vb + (ct * 16 + lo) * 32 + hi * 8;
            bv.q[0] = *(const uint4*)(vr);
            bv.q[1] = *(const uint4*)(vr + 16);
            O[ct] = __builtin_amdgcn_wmma_f32_16x16x32_bf16(
                        false, pa.v, false, bv.v, (short)0, O[ct], false, false);
        }

        if (mb + 1 < NKB) {
            ASYNC_WAIT();          // next block's DMA landed in LDS
            __syncthreads();       // all waves done with `cur`, data visible
        }
    }

    // ---- normalize and store attn[b][n][c] as bf16
    float rinv[8];
    #pragma unroll
    for (int r = 0; r < 8; ++r) rinv[r] = 1.0f / lrow[r];
    #pragma unroll
    for (int ct = 0; ct < 16; ++ct) {
        #pragma unroll
        for (int r = 0; r < 8; ++r) {
            attn[((size_t)(b * NPTS + n0 + r + 8 * hi)) * CMID + ct * 16 + lo] =
                f2bf(O[ct][r] * rinv[r]);
        }
    }
}

// ---------------------------------------------------------------------------
// Kernel 3: final 1x1 conv (Wf 128x256) + BN + ReLU, bf16 in -> f32 out.
// Grid: B * N/16 workgroups x 256 threads; thread = (co = t&127, n-half).
// ---------------------------------------------------------------------------
__global__ __launch_bounds__(256) void final_proj_kernel(
    const unsigned short* __restrict__ attn,
    const float* __restrict__ Wf, const float* __restrict__ bnf,
    float* __restrict__ out)
{
    __shared__ __align__(16) unsigned short al[16 * CMID];  // 8 KB tile
    const int b  = blockIdx.x >> 7;
    const int n0 = (blockIdx.x & 127) * 16;
    const int t  = threadIdx.x;

    {
        int row = t >> 4, col0 = (t & 15) * 16;
        const uint4* src =
            (const uint4*)(attn + ((size_t)(b * NPTS + n0 + row)) * CMID + col0);
        uint4* dst = (uint4*)(al + row * CMID + col0);
        dst[0] = src[0];
        dst[1] = src[1];
    }
    __syncthreads();

    const int co = t & 127;
    const int nh = t >> 7;            // 0/1 -> rows nh*8 .. nh*8+7
    float acc[8];
    #pragma unroll
    for (int j = 0; j < 8; ++j) acc[j] = 0.f;

    union { uint4 q; unsigned short h[8]; } raw;
    for (int c0 = 0; c0 < CMID; c0 += 8) {
        float w[8];
        #pragma unroll
        for (int i = 0; i < 8; ++i) w[i] = Wf[co * CMID + c0 + i];
        #pragma unroll
        for (int j = 0; j < 8; ++j) {
            raw.q = *(const uint4*)(al + (nh * 8 + j) * CMID + c0);
            #pragma unroll
            for (int i = 0; i < 8; ++i)
                acc[j] = fmaf(w[i], bf2f(raw.h[i]), acc[j]);
        }
    }

    float g  = bnf[co];
    float be = bnf[COUT + co];
    float mu = bnf[2 * COUT + co];
    float va = bnf[3 * COUT + co];
    float sc = g * rsqrtf(va + EPSV);
    #pragma unroll
    for (int j = 0; j < 8; ++j) {
        float y = (acc[j] - mu) * sc + be;
        out[((size_t)(b * COUT + co)) * NPTS + n0 + nh * 8 + j] = fmaxf(y, 0.f);
    }
}

// ---------------------------------------------------------------------------
// Host launcher
// ---------------------------------------------------------------------------
extern "C" void kernel_launch(void* const* d_in, const int* in_sizes, int n_in,
                              void* d_out, int out_size, void* d_ws, size_t ws_size,
                              hipStream_t stream)
{
    const float* x   = (const float*)d_in[0];
    const float* Wq  = (const float*)d_in[1];
    const float* Wk  = (const float*)d_in[2];
    const float* Wv  = (const float*)d_in[3];
    const float* Wf  = (const float*)d_in[4];
    const float* bnq = (const float*)d_in[5];
    const float* bnk = (const float*)d_in[6];
    const float* bnv = (const float*)d_in[7];
    const float* bnf = (const float*)d_in[8];
    float* out = (float*)d_out;

    // workspace: 4 bf16 tensors of B*N*256 = 16 MB each (64 MB total)
    const size_t tensor_elems = (size_t)BSZ * NPTS * CMID;
    unsigned short* qt   = (unsigned short*)d_ws;
    unsigned short* kt   = qt + tensor_elems;
    unsigned short* vt   = kt + tensor_elems;
    unsigned short* attn = vt + tensor_elems;

    qkv_proj_kernel<<<dim3(BSZ * NPTS / 16), dim3(256), 0, stream>>>(
        x, Wq, Wk, Wv, bnq, bnk, bnv, qt, kt, vt);

    flash_attn_kernel<<<dim3(BSZ * 16), dim3(256), 0, stream>>>(
        qt, kt, vt, attn);

    final_proj_kernel<<<dim3(BSZ * NPTS / 16), dim3(256), 0, stream>>>(
        attn, Wf, bnf, out);
}
